// ScaledDotProductAttention_33681133535677
// MI455X (gfx1250) — matrix operand order repaired and checked by reading, not compiled, not run on hardware
//
#include <hip/hip_runtime.h>
#include <hip/hip_bf16.h>

// Flash attention for B=2,H=12,S=2048,D=64 (fp32 in/out) on gfx1250.
// 64-key blocks per online-softmax update; f16 WMMA (v_wmma_f32_16x16x32_f16)
// for QK^T, P*V, and P row-sums (P x ones); fp32 accumulation; log2-domain
// softmax (native v_exp_f32); K/V staged to LDS as f16 (v_cvt_pk_rtz_f16_f32,
// ds_store_b128) via a two-phase register pipeline; 4 waves (wave32) per WG.

typedef _Float16 half_t;
typedef __attribute__((ext_vector_type(16))) _Float16 v16h;
typedef __attribute__((ext_vector_type(8)))  float    v8f;

#define WMMA_F32_16x16x32_F16(A, B, C) \
    __builtin_amdgcn_wmma_f32_16x16x32_f16(false, (A), false, (B), (short)0, (C), false, false)

constexpr int S_LEN = 2048;
constexpr int HEADS = 12;
constexpr int D_DIM = 64;
constexpr int KB    = 64;           // keys per softmax update (2 staged halves of 32)
constexpr int QT    = 16;           // query rows per wave
constexpr int WAVES = 4;            // waves per workgroup
constexpr int LDK   = D_DIM + 8;    // padded LDS stride in halfs; 144B rows stay 16B-aligned
constexpr int LDP   = KB + 8;       // padded LDS stride (halfs) for P scratch

static __device__ __forceinline__ unsigned pk_f16(float a, float b) {
    auto h = __builtin_amdgcn_cvt_pkrtz(a, b);     // v_cvt_pk_rtz_f16_f32
    return __builtin_bit_cast(unsigned, h);
}

__global__ __launch_bounds__(WAVES * 32)
void fa_fwd_kernel(const float* __restrict__ q,
                   const float* __restrict__ k,
                   const float* __restrict__ v,
                   const int*   __restrict__ mask,
                   float*       __restrict__ out)
{
    __shared__ _Float16 ldsK[KB][LDK];
    __shared__ _Float16 ldsV[KB][LDK];
    __shared__ _Float16 ldsP[WAVES][QT][LDP];

    const int tid  = threadIdx.x;
    const int lane = tid & 31;
    const int wave = tid >> 5;
    const int half = (lane >> 4) & 1;   // which 16-lane group
    const int l16  = lane & 15;

    constexpr int QBLKS = S_LEN / (QT * WAVES);        // 32 q-blocks per (b,h)
    const int bh    = blockIdx.x / QBLKS;              // 0..23
    const int qblk  = blockIdx.x % QBLKS;
    const int b     = bh / HEADS;
    const int qbase = qblk * (QT * WAVES) + wave * QT; // this wave's 16 q rows

    const float* Qp = q + (size_t)bh * S_LEN * D_DIM;
    const float* Kp = k + (size_t)bh * S_LEN * D_DIM;
    const float* Vp = v + (size_t)bh * S_LEN * D_DIM;
    const int*   Mp = mask + (size_t)b * S_LEN * S_LEN;
    float*       Op = out + (size_t)bh * S_LEN * D_DIM;

    // Softmax scale folded into Q, in log2 domain: exp(s*scale) == exp2(s*scale*log2e)
    const float qscale = 0.125f * 1.44269504088896340736f;

    // ---- Q fragments (A operand 16x32, in this kernel's own k order; B matches): row M = l16 per lane,
//      K contiguous over VGPRs: v0..7 -> K {0..15}+half*16.
    v16h aq0{}, aq1{};   // d = 0..31 and d = 32..63
    {
        const float* qrow = Qp + (size_t)(qbase + l16) * D_DIM;
        #pragma unroll
        for (int vv = 0; vv < 8; ++vv) {
            const int k0 = half * 16 + vv * 2;
            aq0[2*vv]   = (_Float16)(qrow[k0]          * qscale);
            aq0[2*vv+1] = (_Float16)(qrow[k0 + 1]      * qscale);
            aq1[2*vv]   = (_Float16)(qrow[32 + k0]     * qscale);
            aq1[2*vv+1] = (_Float16)(qrow[32 + k0 + 1] * qscale);
        }
    }

    // Constant all-ones B fragment for row-sum WMMA (rowsum(P) = P x ones)
    v16h bones;
    #pragma unroll
    for (int i = 0; i < 16; ++i) bones[i] = (_Float16)1.0f;

    // Output accumulators: 16x64 f32 as four 16x16 C/D tiles (row M = r + half*8, col = dc*16 + l16)
    v8f o0{}, o1{}, o2{}, o3{};
    float mrow[8], lrow[8];
    #pragma unroll
    for (int r = 0; r < 8; ++r) { mrow[r] = -3.0e38f; lrow[r] = 0.0f; }

    // ---- Two-phase register pipeline: 128 threads x 16 floats per matrix per 32-key half.
    const int srow = tid >> 2;            // 0..31 (key row within half-block)
    const int scb  = (tid & 3) * 16;      // 0,16,32,48 (d column base)
    float4 kr0, kr1, kr2, kr3, vr0, vr1, vr2, vr3;
    {   // prologue: rows 0..31 of block 0
        const float4* kg = (const float4*)(Kp + (size_t)srow * D_DIM + scb);
        const float4* vg = (const float4*)(Vp + (size_t)srow * D_DIM + scb);
        kr0 = kg[0]; kr1 = kg[1]; kr2 = kg[2]; kr3 = kg[3];
        vr0 = vg[0]; vr1 = vg[1]; vr2 = vg[2]; vr3 = vg[3];
    }

    for (int kb = 0; kb < S_LEN; kb += KB) {
        // ================= phase A: stage keys kb..kb+31 =================
        __syncthreads();   // all waves done reading LDS from previous iteration
        {
            uint4 pa, pb;
            pa.x = pk_f16(kr0.x, kr0.y); pa.y = pk_f16(kr0.z, kr0.w);
            pa.z = pk_f16(kr1.x, kr1.y); pa.w = pk_f16(kr1.z, kr1.w);
            pb.x = pk_f16(kr2.x, kr2.y); pb.y = pk_f16(kr2.z, kr2.w);
            pb.z = pk_f16(kr3.x, kr3.y); pb.w = pk_f16(kr3.z, kr3.w);
            *(uint4*)&ldsK[srow][scb]     = pa;
            *(uint4*)&ldsK[srow][scb + 8] = pb;
            pa.x = pk_f16(vr0.x, vr0.y); pa.y = pk_f16(vr0.z, vr0.w);
            pa.z = pk_f16(vr1.x, vr1.y); pa.w = pk_f16(vr1.z, vr1.w);
            pb.x = pk_f16(vr2.x, vr2.y); pb.y = pk_f16(vr2.z, vr2.w);
            pb.z = pk_f16(vr3.x, vr3.y); pb.w = pk_f16(vr3.z, vr3.w);
            *(uint4*)&ldsV[srow][scb]     = pa;
            *(uint4*)&ldsV[srow][scb + 8] = pb;
        }
        {   // issue loads for keys kb+32..kb+63 (consumed after s0/s1 compute)
            const float4* kg = (const float4*)(Kp + (size_t)(kb + 32 + srow) * D_DIM + scb);
            const float4* vg = (const float4*)(Vp + (size_t)(kb + 32 + srow) * D_DIM + scb);
            kr0 = kg[0]; kr1 = kg[1]; kr2 = kg[2]; kr3 = kg[3];
            vr0 = vg[0]; vr1 = vg[1]; vr2 = vg[2]; vr3 = vg[3];
        }
        __syncthreads();

        // ---- Scores for keys 0..31 of the block
        v8f s0{}, s1{};
        {
            v16h bk00{}, bk01{}, bk10{}, bk11{};
            #pragma unroll
            for (int vv = 0; vv < 8; ++vv) {
                const int d0 = half * 16 + 2 * vv;
                bk00[2*vv] = ldsK[l16][d0];           bk00[2*vv+1] = ldsK[l16][d0 + 1];
                bk01[2*vv] = ldsK[l16][32 + d0];      bk01[2*vv+1] = ldsK[l16][32 + d0 + 1];
                bk10[2*vv] = ldsK[16 + l16][d0];      bk10[2*vv+1] = ldsK[16 + l16][d0 + 1];
                bk11[2*vv] = ldsK[16 + l16][32 + d0]; bk11[2*vv+1] = ldsK[16 + l16][32 + d0 + 1];
            }
            s0 = WMMA_F32_16x16x32_F16(aq0, bk00, s0);
            s0 = WMMA_F32_16x16x32_F16(aq1, bk01, s0);
            s1 = WMMA_F32_16x16x32_F16(aq0, bk10, s1);
            s1 = WMMA_F32_16x16x32_F16(aq1, bk11, s1);
        }
        #pragma unroll
        for (int r = 0; r < 8; ++r) {
            const size_t qi = (size_t)(qbase + r + half * 8) * S_LEN;
            const int m0 = Mp[qi + kb + l16];
            const int m1 = Mp[qi + kb + 16 + l16];
            s0[r] = m0 ? s0[r] : -1.0e9f;
            s1[r] = m1 ? s1[r] : -1.0e9f;
        }

        // ================= phase B: stage keys kb+32..kb+63 =================
        // (rows 32..63 were last read before this iteration's first barrier; the
        //  s0/s1 fragment reads above touch only rows 0..31 -> no barrier needed here)
        {
            uint4 pa, pb;
            pa.x = pk_f16(kr0.x, kr0.y); pa.y = pk_f16(kr0.z, kr0.w);
            pa.z = pk_f16(kr1.x, kr1.y); pa.w = pk_f16(kr1.z, kr1.w);
            pb.x = pk_f16(kr2.x, kr2.y); pb.y = pk_f16(kr2.z, kr2.w);
            pb.z = pk_f16(kr3.x, kr3.y); pb.w = pk_f16(kr3.z, kr3.w);
            *(uint4*)&ldsK[32 + srow][scb]     = pa;
            *(uint4*)&ldsK[32 + srow][scb + 8] = pb;
            pa.x = pk_f16(vr0.x, vr0.y); pa.y = pk_f16(vr0.z, vr0.w);
            pa.z = pk_f16(vr1.x, vr1.y); pa.w = pk_f16(vr1.z, vr1.w);
            pb.x = pk_f16(vr2.x, vr2.y); pb.y = pk_f16(vr2.z, vr2.w);
            pb.z = pk_f16(vr3.x, vr3.y); pb.w = pk_f16(vr3.z, vr3.w);
            *(uint4*)&ldsV[32 + srow][scb]     = pa;
            *(uint4*)&ldsV[32 + srow][scb + 8] = pb;
        }
        if (kb + KB < S_LEN) {   // issue loads for the next block's first half
            const float4* kg = (const float4*)(Kp + (size_t)(kb + KB + srow) * D_DIM + scb);
            const float4* vg = (const float4*)(Vp + (size_t)(kb + KB + srow) * D_DIM + scb);
            kr0 = kg[0]; kr1 = kg[1]; kr2 = kg[2]; kr3 = kg[3];
            vr0 = vg[0]; vr1 = vg[1]; vr2 = vg[2]; vr3 = vg[3];
        }
        __syncthreads();   // rows 32..63 stores visible to all waves

        // ---- Scores for keys 32..63 of the block
        v8f s2{}, s3{};
        {
            v16h bk00{}, bk01{}, bk10{}, bk11{};
            #pragma unroll
            for (int vv = 0; vv < 8; ++vv) {
                const int d0 = half * 16 + 2 * vv;
                bk00[2*vv] = ldsK[32 + l16][d0];      bk00[2*vv+1] = ldsK[32 + l16][d0 + 1];
                bk01[2*vv] = ldsK[32 + l16][32 + d0]; bk01[2*vv+1] = ldsK[32 + l16][32 + d0 + 1];
                bk10[2*vv] = ldsK[48 + l16][d0];      bk10[2*vv+1] = ldsK[48 + l16][d0 + 1];
                bk11[2*vv] = ldsK[48 + l16][32 + d0]; bk11[2*vv+1] = ldsK[48 + l16][32 + d0 + 1];
            }
            s2 = WMMA_F32_16x16x32_F16(aq0, bk00, s2);
            s2 = WMMA_F32_16x16x32_F16(aq1, bk01, s2);
            s3 = WMMA_F32_16x16x32_F16(aq0, bk10, s3);
            s3 = WMMA_F32_16x16x32_F16(aq1, bk11, s3);
        }
        #pragma unroll
        for (int r = 0; r < 8; ++r) {
            const size_t qi = (size_t)(qbase + r + half * 8) * S_LEN;
            const int m2 = Mp[qi + kb + 32 + l16];
            const int m3 = Mp[qi + kb + 48 + l16];
            s2[r] = m2 ? s2[r] : -1.0e9f;
            s3[r] = m3 ? s3[r] : -1.0e9f;
        }

        // ---- Online softmax over all 64 keys (one max-reduction + one alpha)
        float alpha[8];
        #pragma unroll
        for (int r = 0; r < 8; ++r) {
            float t = fmaxf(fmaxf(s0[r], s1[r]), fmaxf(s2[r], s3[r]));
            t = fmaxf(t, __shfl_xor(t, 1, 32));
            t = fmaxf(t, __shfl_xor(t, 2, 32));
            t = fmaxf(t, __shfl_xor(t, 4, 32));
            t = fmaxf(t, __shfl_xor(t, 8, 32));
            const float mn = fmaxf(mrow[r], t);
            alpha[r] = __builtin_amdgcn_exp2f(mrow[r] - mn);
            mrow[r]  = mn;
            s0[r] = __builtin_amdgcn_exp2f(s0[r] - mn);
            s1[r] = __builtin_amdgcn_exp2f(s1[r] - mn);
            s2[r] = __builtin_amdgcn_exp2f(s2[r] - mn);
            s3[r] = __builtin_amdgcn_exp2f(s3[r] - mn);
        }

        // ---- Transpose P (C layout, 16x64) -> A layout via per-wave LDS scratch
        {
            _Float16 (*pp)[LDP] = ldsP[wave];
            #pragma unroll
            for (int r = 0; r < 8; ++r) {
                pp[r + half * 8][l16]      = (_Float16)s0[r];
                pp[r + half * 8][16 + l16] = (_Float16)s1[r];
                pp[r + half * 8][32 + l16] = (_Float16)s2[r];
                pp[r + half * 8][48 + l16] = (_Float16)s3[r];
            }
            asm volatile("s_wait_dscnt 0" ::: "memory");   // intra-wave LDS ordering
        }
        v16h ap0{}, ap1{};   // keys 0..31 and 32..63
        {
            _Float16 (*pp)[LDP] = ldsP[wave];
            #pragma unroll
            for (int vv = 0; vv < 8; ++vv) {
                const int kr = half * 16 + vv * 2;
                ap0[2*vv]   = pp[l16][kr];
                ap0[2*vv+1] = pp[l16][kr + 1];
                ap1[2*vv]   = pp[l16][32 + kr];
                ap1[2*vv+1] = pp[l16][32 + kr + 1];
            }
        }

        // ---- Row sums via WMMA: rowsum(P) = P x ones; every lane gets its row's sum.
        {
            v8f rs{};
            rs = WMMA_F32_16x16x32_F16(ap0, bones, rs);
            rs = WMMA_F32_16x16x32_F16(ap1, bones, rs);
            #pragma unroll
            for (int r = 0; r < 8; ++r) lrow[r] = lrow[r] * alpha[r] + rs[r];
        }

        // ---- O = O*alpha + P * V  (four 16-wide d chunks; two K=32 WMMAs each)
        #pragma unroll
        for (int dc = 0; dc < 4; ++dc) {
            v16h bv0{}, bv1{};
            #pragma unroll
            for (int vv = 0; vv < 8; ++vv) {
                const int kr = half * 16 + 2 * vv;   // key row within half-block
                bv0[2*vv]   = ldsV[kr][dc * 16 + l16];
                bv0[2*vv+1] = ldsV[kr + 1][dc * 16 + l16];
                bv1[2*vv]   = ldsV[32 + kr][dc * 16 + l16];
                bv1[2*vv+1] = ldsV[32 + kr + 1][dc * 16 + l16];
            }
            v8f& o = (dc == 0) ? o0 : (dc == 1) ? o1 : (dc == 2) ? o2 : o3;
            #pragma unroll
            for (int r = 0; r < 8; ++r) o[r] *= alpha[r];
            o = WMMA_F32_16x16x32_F16(ap0, bv0, o);
            o = WMMA_F32_16x16x32_F16(ap1, bv1, o);
        }
    }

    // ---- Epilogue: normalize by running sum, store fp32
    #pragma unroll
    for (int r = 0; r < 8; ++r) {
        const float inv = 1.0f / lrow[r];
        float* orow = Op + (size_t)(qbase + r + half * 8) * D_DIM;
        orow[ 0 + l16] = o0[r] * inv;
        orow[16 + l16] = o1[r] * inv;
        orow[32 + l16] = o2[r] * inv;
        orow[48 + l16] = o3[r] * inv;
    }
}

extern "C" void kernel_launch(void* const* d_in, const int* in_sizes, int n_in,
                              void* d_out, int out_size, void* d_ws, size_t ws_size,
                              hipStream_t stream) {
    const float* q   = (const float*)d_in[0];
    const float* k   = (const float*)d_in[1];
    const float* v   = (const float*)d_in[2];
    const int*  mask = (const int*) d_in[3];
    float* out = (float*)d_out;

    const int B = 2;
    const int grid = B * HEADS * (S_LEN / (QT * WAVES));   // 768 workgroups
    fa_fwd_kernel<<<grid, WAVES * 32, 0, stream>>>(q, k, v, mask, out);
}
